// TSLTM_82025285419105
// MI455X (gfx1250) — compile-verified, matrix operand order checked by reading, not run on hardware
//
#include <hip/hip_runtime.h>
#include <hip/hip_bf16.h>

// ---------------------------------------------------------------------------
// Time-gated LSTM for MI455X (gfx1250, wave32, WMMA bf16 16x16x32).
//   B=128, S=512, D=H=512, 4H=2048.
// Strategy:
//   1) Hoist x@W out of the scan: one big WMMA GEMM (65536x512 @ 512x2048).
//   2) Per-step fused kernel, split-K for occupancy: 256 blocks x 8 waves,
//      each wave computes a K=64 slice of the 5 GEMM tiles for one 16x16
//      (B,H) tile (c_s = c@W_d plus 4 gate tiles of h@U), partials reduced
//      through LDS, then the sigmoid/tanh gating runs in-register and the
//      h/c state ping-pongs (bf16 for the next GEMM + f32 exact carry).
//   3) Weights pre-converted to transposed bf16 so A and B fragments are
//      two contiguous 16B loads per lane (ISA 16-bit A/B VGPR striping).
// ---------------------------------------------------------------------------

typedef __bf16 bf16_t;
typedef __attribute__((ext_vector_type(16))) __bf16 v16bf;
typedef __attribute__((ext_vector_type(8)))  __bf16 v8bf;
typedef __attribute__((ext_vector_type(8)))  float   v8f;

#define WMMA_BF16(a, b, c) \
  __builtin_amdgcn_wmma_f32_16x16x32_bf16(false, (a), false, (b), (short)0, (c), false, false)

// Problem constants
#define BB   128
#define SS   512
#define HH   512
#define G4   2048
#define KSL  8      // K-slices (waves) per step-kernel block; each covers K=64

// Load one 16x32 bf16 fragment (A layout; B uses the same striping with the
// matrix stored N-major/transposed). r0 = tile row (or col for B), k0 = K base.
static __device__ __forceinline__ v16bf load_frag16(const bf16_t* __restrict__ mat,
                                                    int r0, int k0, int ld) {
  const int lane = threadIdx.x & 31;
  const int r    = r0 + (lane & 15);
  const int kb   = (lane >> 4) << 3;            // lanes 16-31 start at K+8
  const bf16_t* p = mat + (size_t)r * ld + (k0 + kb);
  v8bf lo = *(const v8bf*)(p);                  // K = kb .. kb+7
  v8bf hi = *(const v8bf*)(p + 16);             // K = kb+16 .. kb+23
  v16bf f;
#pragma unroll
  for (int i = 0; i < 8; ++i) { f[i] = lo[i]; f[i + 8] = hi[i]; }
  return f;
}

// --------------------------- prologue kernels ------------------------------

__global__ void cvt_f32_to_bf16_kernel(const float* __restrict__ src,
                                       bf16_t* __restrict__ dst, int n) {
  int i = blockIdx.x * blockDim.x + threadIdx.x;
  if (i < n) dst[i] = (bf16_t)src[i];
}

// src: (K,N) f32 row-major  ->  dst: (N,K) bf16 row-major (transposed)
__global__ void transpose_to_bf16_kernel(const float* __restrict__ src,
                                         bf16_t* __restrict__ dst, int K, int N) {
  int i = blockIdx.x * blockDim.x + threadIdx.x;
  if (i < K * N) {
    int k = i / N, n = i % N;
    dst[(size_t)n * K + k] = (bf16_t)src[i];
  }
}

// GX[r, :] = bf16( X[r,:] @ W + bias ),  r = b*S + t  (65536 x 2048)
__global__ void __launch_bounds__(256)
gemm_x_kernel(const bf16_t* __restrict__ X,   // (65536, 512) bf16
              const bf16_t* __restrict__ WT,  // (2048, 512) bf16 (W transposed)
              const float*  __restrict__ bias,
              bf16_t*       __restrict__ GX) {
  const int wave = threadIdx.x >> 5;
  const int lane = threadIdx.x & 31;
  const int row0 = blockIdx.x * 32 + (wave >> 2) * 16;   // M tile
  const int col0 = blockIdx.y * 64 + (wave &  3) * 16;   // N tile
  v8f acc = {};
  for (int k = 0; k < HH; k += 32) {
    v16bf a = load_frag16(X,  row0, k, HH);
    v16bf b = load_frag16(WT, col0, k, HH);
    acc = WMMA_BF16(a, b, acc);
  }
  const int col   = col0 + (lane & 15);
  const int rbase = row0 + ((lane >> 4) << 3);
  const float bs  = bias[col];
#pragma unroll
  for (int v = 0; v < 8; ++v)
    GX[(size_t)(rbase + v) * G4 + col] = (bf16_t)(acc[v] + bs);
}

// ----------------------------- per-step kernel -----------------------------
// grid (B/16, H/16) = (8, 32) blocks, 256 threads = 8 waves.
// Wave w computes the K-slice [w*64, w*64+64) of all 5 GEMM tiles for this
// block's 16x16 (B,H) tile; partials are reduced through LDS.
__global__ void __launch_bounds__(256)
step_kernel(int t,
            const bf16_t* __restrict__ hbf,  const bf16_t* __restrict__ cbf,
            const float*  __restrict__ cF,
            bf16_t* __restrict__ hbf_n, bf16_t* __restrict__ cbf_n,
            float*  __restrict__ cF_n,
            const bf16_t* __restrict__ UT,   // (2048, 512) bf16 (U transposed)
            const bf16_t* __restrict__ WdT,  // (512, 512)  bf16 (W_d transposed)
            const bf16_t* __restrict__ GX,   // (65536, 2048) bf16 precomputed x@W+bias
            const float*  __restrict__ dts,  // (B, S, H) f32 time deltas
            float* __restrict__ out) {       // (B, S, H) ++ h_t ++ c_t
  // red[matrix][kslice][lane][v] : 5*8*32*8*4 = 40 KB (8 blocks/WGP vs 320 KB)
  __shared__ __align__(32) float red[5][KSL][32][8];

  const int wave = threadIdx.x >> 5;           // k-slice id (0..7)
  const int lane = threadIdx.x & 31;
  const int row0 = blockIdx.x * 16;            // batch tile
  const int col0 = blockIdx.y * 16;            // hidden tile
  const int kbeg = wave * (HH / KSL);          // 64-wide K slice

  v8f acc[5];
#pragma unroll
  for (int m = 0; m < 5; ++m) acc[m] = (v8f){};

#pragma unroll
  for (int k = kbeg; k < kbeg + (HH / KSL); k += 32) {
    v16bf ah = load_frag16(hbf, row0, k, HH);  // shared A for all 4 gates
    v16bf ac = load_frag16(cbf, row0, k, HH);
    acc[0] = WMMA_BF16(ac, load_frag16(WdT, col0, k, HH), acc[0]);
    acc[1] = WMMA_BF16(ah, load_frag16(UT, 0 * HH + col0, k, HH), acc[1]);
    acc[2] = WMMA_BF16(ah, load_frag16(UT, 1 * HH + col0, k, HH), acc[2]);
    acc[3] = WMMA_BF16(ah, load_frag16(UT, 2 * HH + col0, k, HH), acc[3]);
    acc[4] = WMMA_BF16(ah, load_frag16(UT, 3 * HH + col0, k, HH), acc[4]);
  }

  // Spill per-wave partials (f32 C/D layout: VGPR v -> row v or v+8, lane&15 -> col)
#pragma unroll
  for (int m = 0; m < 5; ++m)
    *(v8f*)(&red[m][wave][lane][0]) = acc[m];
  __syncthreads();

  // One thread per output element: tid -> (row_local = tid>>4, col_local = tid&15).
  const int rl = threadIdx.x >> 4;
  const int cl = threadIdx.x & 15;
  const int srcLane = cl + ((rl >= 8) ? 16 : 0);
  const int v = rl & 7;

  float s[5];
#pragma unroll
  for (int m = 0; m < 5; ++m) {
    float acc_s = 0.0f;
#pragma unroll
    for (int w = 0; w < KSL; ++w) acc_s += red[m][w][srcLane][v];
    s[m] = acc_s;
  }

  const int row = row0 + rl;                   // global batch index
  const int col = col0 + cl;                   // global hidden index
  const size_t hc = (size_t)row * HH + col;
  const size_t gr = ((size_t)row * SS + t) * G4;

  const float cs = tanhf(s[0]);
  const float cp = cF[hc];
  const float dt = dts[((size_t)row * SS + t) * HH + col];
  const float cadj = (cp - cs) + cs / __logf(2.7182818284590452f + dt);
  const float it = 1.0f / (1.0f + __expf(-(s[1] + (float)GX[gr + 0 * HH + col])));
  const float ft = 1.0f / (1.0f + __expf(-(s[2] + (float)GX[gr + 1 * HH + col])));
  const float gt = tanhf(            s[3] + (float)GX[gr + 2 * HH + col]);
  const float ot = 1.0f / (1.0f + __expf(-(s[4] + (float)GX[gr + 3 * HH + col])));
  const float cn = ft * cadj + it * gt;
  const float hn = ot * tanhf(cn);

  out[((size_t)row * SS + t) * HH + col] = hn;  // hidden_seq[b, t, :]
  hbf_n[hc] = (bf16_t)hn;
  cbf_n[hc] = (bf16_t)cn;
  cF_n[hc]  = cn;
  if (t == SS - 1) {                            // tail: (h_t, c_t)
    const size_t base = (size_t)BB * SS * HH;
    out[base + hc] = hn;
    out[base + (size_t)BB * HH + hc] = cn;
  }
}

// ------------------------------- launcher ----------------------------------

extern "C" void kernel_launch(void* const* d_in, const int* in_sizes, int n_in,
                              void* d_out, int out_size, void* d_ws, size_t ws_size,
                              hipStream_t stream) {
  const float* inputs = (const float*)d_in[0];   // (B, S, D)
  const float* tdelt  = (const float*)d_in[1];   // (B, S, D)
  const float* W_d    = (const float*)d_in[2];   // (H, H)
  const float* W      = (const float*)d_in[3];   // (D, 4H)
  const float* U      = (const float*)d_in[4];   // (H, 4H)
  const float* bias   = (const float*)d_in[5];   // (4H)
  float* out = (float*)d_out;

  // Workspace carve-out (~342 MB total), 256B aligned slabs.
  char* ws = (char*)d_ws;
  auto carve = [&](size_t bytes) -> char* {
    char* p = ws;
    ws += (bytes + 255) & ~(size_t)255;
    return p;
  };
  const size_t n_x  = (size_t)BB * SS * HH;       // 33,554,432
  bf16_t* Xbf = (bf16_t*)carve(n_x * 2);          //  64 MB
  bf16_t* WT  = (bf16_t*)carve((size_t)G4 * HH * 2);  // 2 MB
  bf16_t* UT  = (bf16_t*)carve((size_t)G4 * HH * 2);  // 2 MB
  bf16_t* WdT = (bf16_t*)carve((size_t)HH * HH * 2);  // 0.5 MB
  bf16_t* GX  = (bf16_t*)carve(n_x * 4 * 2);      // 256 MB (65536 x 2048 bf16)
  bf16_t* hbf[2] = { (bf16_t*)carve((size_t)BB * HH * 2),
                     (bf16_t*)carve((size_t)BB * HH * 2) };
  bf16_t* cbf[2] = { (bf16_t*)carve((size_t)BB * HH * 2),
                     (bf16_t*)carve((size_t)BB * HH * 2) };
  float*  cF[2]  = { (float*)carve((size_t)BB * HH * 4),
                     (float*)carve((size_t)BB * HH * 4) };

  // h0 = c0 = 0 (memsetAsync is graph-capturable).
  hipMemsetAsync(hbf[0], 0, (size_t)BB * HH * 2, stream);
  hipMemsetAsync(cbf[0], 0, (size_t)BB * HH * 2, stream);
  hipMemsetAsync(cF[0],  0, (size_t)BB * HH * 4, stream);

  // Prologue: convert inputs, transpose+convert weights.
  cvt_f32_to_bf16_kernel<<<(int)((n_x + 255) / 256), 256, 0, stream>>>(inputs, Xbf, (int)n_x);
  transpose_to_bf16_kernel<<<(HH * G4 + 255) / 256, 256, 0, stream>>>(W,   WT,  HH, G4);
  transpose_to_bf16_kernel<<<(HH * G4 + 255) / 256, 256, 0, stream>>>(U,   UT,  HH, G4);
  transpose_to_bf16_kernel<<<(HH * HH + 255) / 256, 256, 0, stream>>>(W_d, WdT, HH, HH);

  // Big parallel GEMM: GX = X @ W + bias  (65536 x 2048).
  gemm_x_kernel<<<dim3((BB * SS) / 32, G4 / 64), 256, 0, stream>>>(Xbf, WT, bias, GX);

  // Sequential scan: one fused split-K kernel per timestep
  // (stream order provides the recurrence dependency).
  for (int t = 0; t < SS; ++t) {
    const int cur = t & 1, nxt = (t + 1) & 1;
    step_kernel<<<dim3(BB / 16, HH / 16), 256, 0, stream>>>(
        t, hbf[cur], cbf[cur], cF[cur], hbf[nxt], cbf[nxt], cF[nxt],
        UT, WdT, GX, tdelt, out);
  }
}
